// AttentionPool2d_28656021799203
// MI455X (gfx1250) — compile-verified, matrix operand order checked by reading, not run on hardware
//
#include <hip/hip_runtime.h>
#include <stdint.h>

// ---- types for CDNA5 WMMA -------------------------------------------------
typedef __attribute__((ext_vector_type(16))) __bf16        v16bf;
typedef __attribute__((ext_vector_type(8)))  float         v8f;
typedef __attribute__((ext_vector_type(4)))  unsigned int  u32x4;

union FragAB { v16bf v; u32x4 q[2]; };

#define BM 128
#define BN 128
#define BK 32
#define LDA 40   // padded row length (bf16 elems): 80B stride breaks bank conflicts

// One async DMA: 32 lanes x 16B -> 512B global->LDS, tracked by ASYNCcnt.
#define ASYNC_LD_B128(ldsoff, gaddr)                                   \
    asm volatile("global_load_async_to_lds_b128 %0, %1, off"           \
                 :: "v"(ldsoff), "v"(gaddr) : "memory")

// ---------------------------------------------------------------------------
// GEMM: C[M,N] = A[M,K] * B[N,K]^T + bias[N]   (A,B bf16 row-major, f32 accum)
// 256 threads = 8 waves; wave grid 2x4 -> each wave computes 64x32 of C
// (4x2 tiles of 16x16 -> 8 v_wmma_f32_16x16x32_bf16 per K-step).
// Double-buffered LDS fed by GLOBAL_LOAD_ASYNC_TO_LDS_B128: next K-tile's
// DMA overlaps current K-tile's WMMAs; s_wait_asynccnt 4 retires the current
// buffer (async loads complete in order).
// ---------------------------------------------------------------------------
template<bool OUT_BF16>
__global__ __launch_bounds__(256)
void gemm_bf16_nt(const __bf16* __restrict__ A,
                  const __bf16* __restrict__ Bm,
                  const float*  __restrict__ bias,
                  float*  __restrict__ Cf,
                  __bf16* __restrict__ Cb,
                  int M, int N, int K)
{
    __shared__ __align__(16) __bf16 ldsA[2][BM * LDA];
    __shared__ __align__(16) __bf16 ldsB[2][BN * LDA];

    const int tid  = threadIdx.x;
    const int lane = tid & 31;
    const int wave = tid >> 5;
    const int wm   = wave >> 2;   // 0..1 : 64-row slab
    const int wn   = wave & 3;    // 0..3 : 32-col slab
    const int hh   = lane >> 4;   // half-wave select (K interleave)
    const int l    = lane & 15;

    const long rowBase = (long)blockIdx.y * BM;
    const long colBase = (long)blockIdx.x * BN;

    // Staging geometry: 512 16B chunks per tile, 2 per thread (ch = tid, tid+256).
    const int r0 = tid >> 2,         sg0 = tid & 3;
    const int r1 = (tid + 256) >> 2, sg1 = (tid + 256) & 3;

    // LDS byte offsets (low 32 bits of generic shared pointer == LDS offset).
    const unsigned a0 = (unsigned)(uintptr_t)&ldsA[0][r0 * LDA + sg0 * 8];
    const unsigned a1 = (unsigned)(uintptr_t)&ldsA[0][r1 * LDA + sg1 * 8];
    const unsigned b0 = (unsigned)(uintptr_t)&ldsB[0][r0 * LDA + sg0 * 8];
    const unsigned b1 = (unsigned)(uintptr_t)&ldsB[0][r1 * LDA + sg1 * 8];
    const unsigned bufA = (unsigned)(BM * LDA * 2);   // bytes ldsA[0]->ldsA[1]
    const unsigned bufB = (unsigned)(BN * LDA * 2);

    // Per-thread global base addresses for the two chunks of A and B.
    const uint64_t gA0 = (uint64_t)(uintptr_t)&A [(rowBase + r0) * (long)K + sg0 * 8];
    const uint64_t gA1 = (uint64_t)(uintptr_t)&A [(rowBase + r1) * (long)K + sg1 * 8];
    const uint64_t gB0 = (uint64_t)(uintptr_t)&Bm[(colBase + r0) * (long)K + sg0 * 8];
    const uint64_t gB1 = (uint64_t)(uintptr_t)&Bm[(colBase + r1) * (long)K + sg1 * 8];

    v8f acc[4][2];
    #pragma unroll
    for (int i = 0; i < 4; ++i)
        #pragma unroll
        for (int j = 0; j < 2; ++j)
            acc[i][j] = v8f{};

    const int steps = K / BK;

    // Prologue: DMA first K-tile into buffer 0.  (ASYNCcnt = 4)
    ASYNC_LD_B128(a0, gA0);
    ASYNC_LD_B128(a1, gA1);
    ASYNC_LD_B128(b0, gB0);
    ASYNC_LD_B128(b1, gB1);

    for (int ks = 0; ks < steps; ++ks) {
        const unsigned cur = (unsigned)(ks & 1);

        if (ks + 1 < steps) {
            // Issue next tile's DMA into the other buffer, then wait for the
            // current tile only (in-order completion => <=4 outstanding).
            const uint64_t kb = (uint64_t)(ks + 1) * (BK * 2);  // bytes
            const unsigned nb = (cur ^ 1u);
            ASYNC_LD_B128(a0 + nb * bufA, gA0 + kb);
            ASYNC_LD_B128(a1 + nb * bufA, gA1 + kb);
            ASYNC_LD_B128(b0 + nb * bufB, gB0 + kb);
            ASYNC_LD_B128(b1 + nb * bufB, gB1 + kb);
            asm volatile("s_wait_asynccnt 0x4" ::: "memory");
        } else {
            asm volatile("s_wait_asynccnt 0x0" ::: "memory");
        }
        __syncthreads();   // whole tile resident (every wave waited on its own DMAs)

        // Per-lane fragment gather per ISA 16-bit A/B layout:
        // lane<16 -> K 0..7 and 16..23 ; lane>=16 -> K 8..15 and 24..31.
        FragAB fa[4], fb[2];
        #pragma unroll
        for (int mt = 0; mt < 4; ++mt) {
            const u32x4* pr = (const u32x4*)&ldsA[cur][(wm * 64 + mt * 16 + l) * LDA];
            fa[mt].q[0] = pr[hh];
            fa[mt].q[1] = pr[hh + 2];
        }
        #pragma unroll
        for (int nt = 0; nt < 2; ++nt) {
            const u32x4* pr = (const u32x4*)&ldsB[cur][(wn * 32 + nt * 16 + l) * LDA];
            fb[nt].q[0] = pr[hh];
            fb[nt].q[1] = pr[hh + 2];
        }

        #pragma unroll
        for (int mt = 0; mt < 4; ++mt)
            #pragma unroll
            for (int nt = 0; nt < 2; ++nt)
                acc[mt][nt] = __builtin_amdgcn_wmma_f32_16x16x32_bf16(
                    false, fa[mt].v, false, fb[nt].v,
                    (short)0, acc[mt][nt], false, false);

        __syncthreads();   // all waves done reading before next DMA overwrites
    }

    // Epilogue: C/D layout -> lane l = column, hh*8+r = row within tile.
    #pragma unroll
    for (int mt = 0; mt < 4; ++mt) {
        const long gm = rowBase + wm * 64 + mt * 16 + hh * 8;
        #pragma unroll
        for (int nt = 0; nt < 2; ++nt) {
            const long gn = colBase + wn * 32 + nt * 16 + l;
            const float bsv = bias[gn];
            #pragma unroll
            for (int r = 0; r < 8; ++r) {
                const float val = acc[mt][nt][r] + bsv;
                if (OUT_BF16) Cb[(gm + r) * (long)N + gn] = (__bf16)val;
                else          Cf[(gm + r) * (long)N + gn] = val;
            }
        }
    }
}

// ---------------------------------------------------------------------------
// Build t[B,50,C] in bf16: token0 = spatial mean + pos[0], token n = x + pos[n]
// ---------------------------------------------------------------------------
__global__ __launch_bounds__(256)
void build_t(const float* __restrict__ x, const float* __restrict__ pos,
             __bf16* __restrict__ t)
{
    const int idx = blockIdx.x * 256 + threadIdx.x;   // over B*C = 524288
    const int b = idx >> 11;
    const int c = idx & 2047;
    const float* xp = x + (long)idx * 49;             // (b*C+c)*49
    float sum = 0.f;
    #pragma unroll
    for (int i = 0; i < 49; ++i) sum += xp[i];
    const long tb = (long)b * 50 * 2048 + c;
    t[tb] = (__bf16)(sum * (1.0f / 49.0f) + pos[c]);
    #pragma unroll
    for (int n = 1; n < 50; ++n)
        t[tb + (long)n * 2048] = (__bf16)(xp[n - 1] + pos[n * 2048 + c]);
}

// ---------------------------------------------------------------------------
// Weight downcast to bf16.
// ---------------------------------------------------------------------------
__global__ __launch_bounds__(256)
void cvt_weights(const float* __restrict__ Wv, const float* __restrict__ Wc,
                 __bf16* __restrict__ wv, __bf16* __restrict__ wc)
{
    const int i = blockIdx.x * 256 + threadIdx.x;     // covers 2048*2048
    wv[i] = (__bf16)Wv[i];
    if (i < 1024 * 2048) wc[i] = (__bf16)Wc[i];
}

// ---------------------------------------------------------------------------
// Attention, query 0 only. One wave32 per (b,h); head dim 64 -> 2 floats/lane.
// Overwrites v row (b*50+0) head-slice in place with ctx0 (read-before-write,
// slices disjoint across waves).
// ---------------------------------------------------------------------------
__global__ __launch_bounds__(256)
void attn_q0(__bf16* __restrict__ v)   // [B*50, 2048]
{
    const int wave = threadIdx.x >> 5;
    const int lane = threadIdx.x & 31;
    const int pair = blockIdx.x * 8 + wave;   // 0..8191
    const int b = pair >> 5;                  // /32 heads
    const int h = pair & 31;
    const long rowBase = (long)b * 50 * 2048;
    const long col     = (long)h * 64 + lane * 2;

    const float q0 = (float)v[rowBase + col];
    const float q1 = (float)v[rowBase + col + 1];

    float s[50];
    float mx = -1e30f;
    #pragma unroll
    for (int m = 0; m < 50; ++m) {
        const long off = rowBase + (long)m * 2048 + col;
        float p = q0 * (float)v[off] + q1 * (float)v[off + 1];
        #pragma unroll
        for (int o = 16; o > 0; o >>= 1) p += __shfl_xor(p, o, 32);
        p *= 0.125f;                           // hd^-0.5, hd=64
        s[m] = p;
        mx = fmaxf(mx, p);
    }
    float den = 0.f;
    #pragma unroll
    for (int m = 0; m < 50; ++m) { s[m] = __expf(s[m] - mx); den += s[m]; }
    const float inv = 1.0f / den;

    float c0 = 0.f, c1 = 0.f;
    #pragma unroll
    for (int m = 0; m < 50; ++m) {
        const long off = rowBase + (long)m * 2048 + col;
        c0 += s[m] * (float)v[off];
        c1 += s[m] * (float)v[off + 1];
    }
    v[rowBase + col]     = (__bf16)(c0 * inv);
    v[rowBase + col + 1] = (__bf16)(c1 * inv);
}

// ---------------------------------------------------------------------------
extern "C" void kernel_launch(void* const* d_in, const int* in_sizes, int n_in,
                              void* d_out, int out_size, void* d_ws, size_t ws_size,
                              hipStream_t stream)
{
    const float* x   = (const float*)d_in[0];  // [256,2048,7,7]
    const float* pos = (const float*)d_in[1];  // [50,2048]
    const float* Wv  = (const float*)d_in[2];  // [2048,2048]
    const float* bv  = (const float*)d_in[3];  // [2048]
    const float* Wc  = (const float*)d_in[4];  // [1024,2048]
    const float* bc  = (const float*)d_in[5];  // [1024]
    float* out = (float*)d_out;                // [256,50,1024]

    char* ws = (char*)d_ws;
    __bf16* t  = (__bf16*)ws;  ws += (size_t)12800 * 2048 * 2;  // 52.4 MB
    __bf16* v  = (__bf16*)ws;  ws += (size_t)12800 * 2048 * 2;  // 52.4 MB
    __bf16* wv = (__bf16*)ws;  ws += (size_t)2048 * 2048 * 2;   //  8.4 MB
    __bf16* wc = (__bf16*)ws;                                   //  4.2 MB

    cvt_weights<<<(2048 * 2048) / 256, 256, 0, stream>>>(Wv, Wc, wv, wc);
    build_t<<<(256 * 2048) / 256, 256, 0, stream>>>(x, pos, t);

    // v = t @ Wv^T + bv  : M=12800, N=2048, K=2048 (bf16 out)
    gemm_bf16_nt<true><<<dim3(2048 / BN, 12800 / BM), 256, 0, stream>>>(
        t, wv, bv, nullptr, v, 12800, 2048, 2048);

    // replace v[:,0,:] with attention context (query 0 only)
    attn_q0<<<(256 * 32) / 8, 256, 0, stream>>>(v);

    // out = v @ Wc^T + bc : M=12800, N=1024, K=2048 (f32 out) — final answer
    gemm_bf16_nt<false><<<dim3(1024 / BN, 12800 / BM), 256, 0, stream>>>(
        v, wc, bc, out, nullptr, 12800, 1024, 2048);
}
